// SelfAttentionLayer_21285857919426
// MI455X (gfx1250) — matrix operand order repaired and checked by reading, not compiled, not run on hardware
//
#include <hip/hip_runtime.h>
#include <hip/hip_bf16.h>
#include <math.h>

// ---------------------------------------------------------------------------
// CDNA5 (gfx1250) wave32 WMMA self-attention:
//   kqv = x @ W_kqv + b_kqv   (bf16 in, f32 acc, v_wmma_f32_16x16x32_bf16)
//   flash attention, online softmax, causal mask, scale = 1/sqrt(T)
//   out = att_out @ W_proj + b_proj (f32 out)
// Fragment moves are 128-bit; GEMM A-tiles staged by the Tensor Data Mover
// (tensor_load_to_lds, TENSORcnt-tracked, double-buffered) when available.
// ---------------------------------------------------------------------------

typedef __attribute__((ext_vector_type(16))) __bf16    v16bf;
typedef __attribute__((ext_vector_type(8)))  float     v8f;
typedef __attribute__((ext_vector_type(4)))  unsigned  u32x4;
typedef __attribute__((ext_vector_type(2)))  unsigned  u32x2;
typedef __attribute__((ext_vector_type(4)))  float     f32x4;
typedef __attribute__((ext_vector_type(8)))  int       i32x8;
typedef __attribute__((ext_vector_type(4)))  int       i32x4;

#if __has_builtin(__builtin_amdgcn_tensor_load_to_lds) && \
    __has_builtin(__builtin_amdgcn_s_wait_tensorcnt)
#define USE_TDM 1
#else
#define USE_TDM 0
#endif

union FragBF {
  v16bf v;
  u32x4 q[2];
};

__device__ __forceinline__ v8f v8f_zero() {
  v8f z;
#pragma unroll
  for (int i = 0; i < 8; ++i) z[i] = 0.0f;
  return z;
}

__device__ __forceinline__ unsigned short f32_to_bf16_rne(float f) {
  unsigned u = __float_as_uint(f);
  unsigned r = u + 0x7FFFu + ((u >> 16) & 1u);
  return (unsigned short)(r >> 16);
}

// bf16 A-matrix 16x32 (ISA 7.12.2): per lane, two contiguous 16B runs at byte
// offsets {16*half, 32+16*half} of the row -> ushort offsets {8h, 16+8h}.
// bf16 B-matrix 32x16: two 16B runs at bytes {16*half, 32+16*half} of the transposed column
// -> ushort offsets {8h, 16+8h}.

// LDS row stride for 32 bf16 + pad: 40 ushorts = 80B (16B aligned, 20-dword
// stride -> conflict-free bank rotation mod 64).
#define LDSS 40

#if USE_TDM
// ---------------------------------------------------------------------------
// TDM: async DMA of a [rows x 32] bf16 tile (row stride = K elements) from
// global to LDS, padding each 64B row with 4 dwords -> LDSS stride in LDS.
// D# packing per CDNA5 ISA:
//   g0: count=1 | lds_addr | global_addr | type=2
//   g1: data_size=1(2B), pad_enable, pad_interval=3(16 dwords),
//       pad_amount=3(4 dwords), tensor_dim0=K, tensor_dim1=rows,
//       tile_dim0=32, tile_dim1=rows, tensor_dim0_stride=K
// ---------------------------------------------------------------------------
__device__ __forceinline__ void tdm_load_tile_a(const unsigned short* gsrc,
                                                unsigned lds_byte, int K,
                                                int rows) {
  unsigned long long ga = (unsigned long long)(uintptr_t)gsrc;
  u32x4 g0;
  g0[0] = 1u;                                   // count=1, user descriptor
  g0[1] = lds_byte;                             // LDS destination (bytes)
  g0[2] = (unsigned)ga;                         // global_addr[31:0]
  g0[3] = (unsigned)(ga >> 32) | 0x80000000u;   // global_addr[56:32]|type=2
  i32x8 g1;
  g1[0] = (1 << 16) | (1 << 20) | (3 << 22) | (3 << 25);
  g1[1] = (int)((unsigned)(K & 0xffff) << 16);            // tensor_dim0 lo16
  g1[2] = (int)(((unsigned)rows << 16) | ((K >> 16) & 0xffff));
  g1[3] = (int)(32u << 16);                               // tile_dim0 = 32
  g1[4] = rows;                                           // tile_dim1
  g1[5] = K;                                              // dim0_stride lo32
  g1[6] = 0;
  g1[7] = 0;
  i32x4 z4;
  z4[0] = z4[1] = z4[2] = z4[3] = 0;
#if defined(__clang_major__) && (__clang_major__ >= 23)
  i32x8 z8;
#pragma unroll
  for (int i = 0; i < 8; ++i) z8[i] = 0;
  __builtin_amdgcn_tensor_load_to_lds(g0, g1, z4, z4, z8, 0);
#else
  __builtin_amdgcn_tensor_load_to_lds(g0, g1, z4, z4, 0);
#endif
}
#endif  // USE_TDM

// ---------------------------------------------------------------------------
// fp32 -> bf16 conversion (float4 in, dword2 out). n % 4 == 0.
// ---------------------------------------------------------------------------
__global__ void cvt_f32_to_bf16(const float* __restrict__ src,
                                unsigned short* __restrict__ dst, int n) {
  int i = blockIdx.x * blockDim.x + threadIdx.x;
  int stride = gridDim.x * blockDim.x;
  int n4 = n >> 2;
  for (; i < n4; i += stride) {
    f32x4 v = *(const f32x4*)(src + 4 * (size_t)i);
    u32x2 o;
    o[0] = (unsigned)f32_to_bf16_rne(v[0]) | ((unsigned)f32_to_bf16_rne(v[1]) << 16);
    o[1] = (unsigned)f32_to_bf16_rne(v[2]) | ((unsigned)f32_to_bf16_rne(v[3]) << 16);
    *(u32x2*)(dst + 4 * (size_t)i) = o;
  }
}

// ---------------------------------------------------------------------------
// GEMM: out[M,N] = A[M,K](bf16) * B[K,N](bf16) + bias[N], f32 accumulate.
// 128x64 tile per 128-thread (4-wave) workgroup; each wave owns a 32x64 strip
// (2 A-frags x 4 B-frags = 8 WMMAs per 32-deep k-step). A tile staged by TDM
// (double buffered, overlapped with compute); B tile staged transposed via
// b128 loads + register interleave. Requires M%128==0, N%64==0, K%32==0.
// ---------------------------------------------------------------------------
template <bool OUT_BF16>
__global__ __launch_bounds__(128) void gemm_bias_bf16(
    const unsigned short* __restrict__ A, const unsigned short* __restrict__ B,
    const float* __restrict__ bias, void* __restrict__ outp, int M, int N,
    int K) {
  __shared__ unsigned short sA[2 * 128 * LDSS];  // A tile 128x32, dbl buffered
  __shared__ unsigned short sBT[64 * LDSS];      // B tile 32x64, [n][k]

  const int tid = threadIdx.x;
  const int lane = tid & 31;
  const int wv = tid >> 5;          // 0..3
  const int half = (lane >> 4) & 1;
  const int ln = lane & 15;

  const int n0 = blockIdx.x * 64;
  const int m0 = blockIdx.y * 128;

  v8f acc[2][4];
#pragma unroll
  for (int s = 0; s < 2; ++s)
#pragma unroll
    for (int t = 0; t < 4; ++t) acc[s][t] = v8f_zero();

  // B-staging task (one per thread): key-row pair kp, 8-column chunk cc
  const int kp = tid >> 3;
  const int cc = (tid & 7) * 8;
  const unsigned short* bstage = B + (size_t)(2 * kp) * N + n0 + cc;
  unsigned short* bdst = sBT + cc * LDSS + 2 * kp;

  const unsigned short* atile = A + (size_t)m0 * K;  // tile row 0
  constexpr unsigned kBufBytes = 128 * LDSS * 2;

#if USE_TDM
  const unsigned ldsA = (unsigned)(uintptr_t)sA;
  if (tid < 32) tdm_load_tile_a(atile, ldsA, K, 128);  // prologue: buffer 0
#endif

  for (int k0 = 0; k0 < K; k0 += 32) {
    const int cur = (k0 >> 5) & 1;
    unsigned short* sAbuf = sA + cur * 128 * LDSS;

#if !USE_TDM
    // fallback: cooperative A tile load (4 b128 chunks per thread)
    for (int c = tid; c < 512; c += 128) {
      int row = c >> 2;
      int q = c & 3;
      u32x4 val = *(const u32x4*)(atile + (size_t)row * K + k0 + q * 8);
      *(u32x4*)(sAbuf + row * LDSS + q * 8) = val;
    }
#endif

    // ---- B tile 32x64 -> sBT[n][k]: interleave adjacent k rows ----
    {
      u32x4 e = *(const u32x4*)bstage;        // k row 2kp,  cols cc..cc+7
      u32x4 o = *(const u32x4*)(bstage + N);  // k row 2kp+1
      bstage += (size_t)32 * N;
      __builtin_prefetch(bstage, 0, 1);       // next k-step B rows
#pragma unroll
      for (int i = 0; i < 4; ++i) {
        unsigned ew = e[i], ow = o[i];
        unsigned lo = (ew & 0xffffu) | (ow << 16);       // col cc+2i
        unsigned hi = (ew >> 16) | (ow & 0xffff0000u);   // col cc+2i+1
        *(unsigned*)(bdst + (2 * i + 0) * LDSS) = lo;
        *(unsigned*)(bdst + (2 * i + 1) * LDSS) = hi;
      }
    }

#if USE_TDM
    if (tid < 32) {
      __builtin_amdgcn_s_wait_tensorcnt(0);   // current A buffer complete
      if (k0 + 32 < K)                        // kick next A tile -> other buf
        tdm_load_tile_a(atile + k0 + 32, ldsA + (cur ^ 1) * kBufBytes, K, 128);
    }
#endif
    __syncthreads();

    FragBF fa[2];
#pragma unroll
    for (int s = 0; s < 2; ++s) {
      const unsigned short* arow = sAbuf + (32 * wv + 16 * s + ln) * LDSS;
      fa[s].q[0] = *(const u32x4*)(arow + 8 * half);
      fa[s].q[1] = *(const u32x4*)(arow + 16 + 8 * half);
    }
#pragma unroll
    for (int t = 0; t < 4; ++t) {
      FragBF fb;
      const unsigned short* brow = sBT + (t * 16 + ln) * LDSS;
      fb.q[0] = *(const u32x4*)(brow + 8 * half);
      fb.q[1] = *(const u32x4*)(brow + 16 + 8 * half);
#pragma unroll
      for (int s = 0; s < 2; ++s)
        acc[s][t] = __builtin_amdgcn_wmma_f32_16x16x32_bf16(
            false, fa[s].v, false, fb.v, (short)0, acc[s][t], false, false);
    }
    __syncthreads();
  }

  // epilogue: row = m0 + 32*wv + 16*s + r + 8*half, col = n0 + 16*t + ln
#pragma unroll
  for (int t = 0; t < 4; ++t) {
    int col = n0 + t * 16 + ln;
    float bv = bias[col];
#pragma unroll
    for (int s = 0; s < 2; ++s) {
#pragma unroll
      for (int r = 0; r < 8; ++r) {
        int row = m0 + 32 * wv + 16 * s + r + 8 * half;
        float v = acc[s][t][r] + bv;
        if (OUT_BF16)
          ((unsigned short*)outp)[(size_t)row * N + col] = f32_to_bf16_rne(v);
        else
          ((float*)outp)[(size_t)row * N + col] = v;
      }
    }
  }
}

// ---------------------------------------------------------------------------
// Flash attention: one 512-thread (16-wave) workgroup per (batch, 16-row query
// block). Wave w owns output columns [64w, 64w+64). C=1024 split across the
// 16 waves for QK^T; partial 16x32 score tiles reduced via LDS ds_add_f32.
// V block (32 keys x 1024 cols) staged transposed in LDS with b128 loads +
// register interleave so P*V fragments are pure ds_load_b128. All global
// pointers advance by the constant 32*3C per key-block (no muls in the loop).
// kqv layout: [B,T,3C] bf16, split order (k | q | v).
// ---------------------------------------------------------------------------
__global__ __launch_bounds__(512) void attn_flash_bf16(
    const unsigned short* __restrict__ kqv, unsigned short* __restrict__ attout,
    int T, int C) {
  __shared__ unsigned short sVT[1024 * LDSS];  // V^T: [col][key] (80 KB)
  __shared__ float sS[16 * 32];
  __shared__ unsigned short sP[16 * LDSS];
  __shared__ float sM[16], sL[16], sScale[16];

  const int tid = threadIdx.x;
  const int lane = tid & 31;
  const int wv = tid >> 5;          // 0..15
  const int half = (lane >> 4) & 1;
  const int ln = lane & 15;

  const int qb = blockIdx.x;
  const int b = blockIdx.y;
  const int qrow0 = qb * 16;

  const size_t C3 = (size_t)3 * C;
  const size_t stepB = 32 * C3;                         // key-block stride
  const unsigned short* kg = kqv + (size_t)b * T * C3;  // k at +0
  const unsigned short* qg = kg + C;                    // q at +C
  const unsigned short* vg = kg + 2 * (size_t)C;        // v at +2C

  const float scl = rsqrtf((float)T);  // reference scales by 1/sqrt(T)
  const float NEGINF = -__builtin_inff();

  // Q fragments for this wave's 64-wide C slice: 4 b128 loads, live in VGPRs.
  FragBF fq[2];
  {
    const unsigned short* qrowp = qg + (size_t)(qrow0 + ln) * C3;
#pragma unroll
    for (int ks = 0; ks < 2; ++ks) {
      const unsigned short* base = qrowp + 64 * wv + 32 * ks;
      fq[ks].q[0] = *(const u32x4*)(base + 8 * half);
      fq[ks].q[1] = *(const u32x4*)(base + 16 + 8 * half);
    }
  }

  // Running pointers: K fragment source, V staging sources/destinations.
  const unsigned short* kit = kg + (size_t)ln * C3 + 64 * wv;
  const unsigned short* vsrc[4];
  unsigned short* vdst[4];
#pragma unroll
  for (int it = 0; it < 4; ++it) {
    int s = tid + 512 * it;
    int vkp = s >> 7;            // key pair 0..15
    int vcc = (s & 127) * 8;     // column chunk
    vsrc[it] = vg + (size_t)(2 * vkp) * C3 + vcc;
    vdst[it] = sVT + vcc * LDSS + 2 * vkp;
  }

  v8f acc[4];
#pragma unroll
  for (int t = 0; t < 4; ++t) acc[t] = v8f_zero();

  if (tid < 16) {
    sM[tid] = NEGINF;
    sL[tid] = 0.0f;
  }
  __syncthreads();

  const int jlim = (qrow0 + 15) / 32;  // inclusive
  for (int j = 0; j <= jlim; ++j) {
    const int key0 = j * 32;

    sS[tid] = 0.0f;  // 16*32 == blockDim

    // ---- stage V^T: 16 key-pairs x 128 8-col chunks = 2048 tasks, 4/thread
#pragma unroll
    for (int it = 0; it < 4; ++it) {
      u32x4 e = *(const u32x4*)vsrc[it];         // key row 2kp
      u32x4 o = *(const u32x4*)(vsrc[it] + C3);  // key row 2kp+1
      vsrc[it] += stepB;
      __builtin_prefetch(vsrc[it], 0, 1);        // next key block V rows
      unsigned short* d = vdst[it];
#pragma unroll
      for (int i = 0; i < 4; ++i) {
        unsigned ew = e[i], ow = o[i];
        unsigned lo = (ew & 0xffffu) | (ow << 16);
        unsigned hi = (ew >> 16) | (ow & 0xffff0000u);
        *(unsigned*)(d + (2 * i + 0) * LDSS) = lo;
        *(unsigned*)(d + (2 * i + 1) * LDSS) = hi;
      }
    }
    __syncthreads();

    // ---- partial S = Q_slice * K_slice^T over this wave's 64 C-dims ----
#pragma unroll
    for (int kh = 0; kh < 2; ++kh) {
      v8f ps = v8f_zero();
      const unsigned short* krow = kit + (size_t)(16 * kh) * C3;
#pragma unroll
      for (int ks = 0; ks < 2; ++ks) {
        FragBF fk;
        const unsigned short* base = krow + 32 * ks;
        fk.q[0] = *(const u32x4*)(base + 8 * half);
        fk.q[1] = *(const u32x4*)(base + 16 + 8 * half);
        ps = __builtin_amdgcn_wmma_f32_16x16x32_bf16(
            false, fq[ks].v, false, fk.v, (short)0, ps, false, false);
      }
#pragma unroll
      for (int r = 0; r < 8; ++r) {
        int m = r + 8 * half;
        atomicAdd(&sS[m * 32 + 16 * kh + ln], ps[r]);  // ds_add_f32 reduce
      }
    }
    kit += stepB;
    __builtin_prefetch(kit, 0, 1);  // next key block K rows
    __syncthreads();

    // ---- online softmax update, one thread per query row ----
    if (tid < 16) {
      const int m = tid;
      const int qrow = qrow0 + m;
      float mOld = sM[m];
      float mNew = mOld;
      float s[32];
#pragma unroll
      for (int c = 0; c < 32; ++c) {
        float x = sS[m * 32 + c] * scl;
        if (key0 + c > qrow) x = NEGINF;  // causal mask
        s[c] = x;
        mNew = fmaxf(mNew, x);
      }
      float scale = (mOld == NEGINF) ? 0.0f : __expf(mOld - mNew);
      float lsum = 0.0f;
#pragma unroll
      for (int c = 0; c < 32; ++c) {
        float p = (s[c] == NEGINF) ? 0.0f : __expf(s[c] - mNew);
        sP[m * LDSS + c] = f32_to_bf16_rne(p);
        lsum += p;
      }
      sM[m] = mNew;
      sL[m] = sL[m] * scale + lsum;
      sScale[m] = scale;
    }
    __syncthreads();

    // ---- rescale accumulators, accumulate P(16x32) * V(32x64 slice) ----
#pragma unroll
    for (int t = 0; t < 4; ++t)
#pragma unroll
      for (int r = 0; r < 8; ++r) acc[t][r] *= sScale[r + 8 * half];

    FragBF fp;
    {
      const unsigned short* prow = sP + ln * LDSS;
      fp.q[0] = *(const u32x4*)(prow + 8 * half);
      fp.q[1] = *(const u32x4*)(prow + 16 + 8 * half);
    }
#pragma unroll
    for (int t = 0; t < 4; ++t) {
      FragBF fv;
      const unsigned short* vcol = sVT + (64 * wv + t * 16 + ln) * LDSS;
      fv.q[0] = *(const u32x4*)(vcol + 8 * half);
      fv.q[1] = *(const u32x4*)(vcol + 16 + 8 * half);
      acc[t] = __builtin_amdgcn_wmma_f32_16x16x32_bf16(
          false, fp.v, false, fv.v, (short)0, acc[t], false, false);
    }
    __syncthreads();
  }

  // ---- finalize: divide by l, store bf16 for projection GEMM ----
#pragma unroll
  for (int t = 0; t < 4; ++t) {
    const int col = 64 * wv + t * 16 + ln;
#pragma unroll
    for (int r = 0; r < 8; ++r) {
      int m = r + 8 * half;
      float inv = 1.0f / sL[m];
      attout[(size_t)(b * T + qrow0 + m) * C + col] =
          f32_to_bf16_rne(acc[t][r] * inv);
    }
  }
}

// ---------------------------------------------------------------------------
// Launch
// ---------------------------------------------------------------------------
extern "C" void kernel_launch(void* const* d_in, const int* in_sizes, int n_in,
                              void* d_out, int out_size, void* d_ws,
                              size_t ws_size, hipStream_t stream) {
  (void)in_sizes; (void)n_in; (void)out_size; (void)ws_size;

  const float* x      = (const float*)d_in[0];
  const float* W_kqv  = (const float*)d_in[1];
  const float* b_kqv  = (const float*)d_in[2];
  const float* W_proj = (const float*)d_in[3];
  const float* b_proj = (const float*)d_in[4];

  const int B = 8, T = 2048, C = 1024;
  const int M = B * T;    // 16384
  const int N1 = 3 * C;   // 3072

  // workspace layout (bytes, all 16B-aligned)
  char* ws = (char*)d_ws;
  unsigned short* x_bf     = (unsigned short*)(ws + 0);           //  32 MB
  unsigned short* wkqv_bf  = (unsigned short*)(ws + 33554432);    //   6 MB
  unsigned short* wproj_bf = (unsigned short*)(ws + 39845888);    //   2 MB
  unsigned short* kqv_bf   = (unsigned short*)(ws + 41943040);    //  96 MB
  unsigned short* ao_bf    = (unsigned short*)(ws + 142606336);   //  32 MB

  cvt_f32_to_bf16<<<1024, 256, 0, stream>>>(x, x_bf, M * C);
  cvt_f32_to_bf16<<<1024, 256, 0, stream>>>(W_kqv, wkqv_bf, C * N1);
  cvt_f32_to_bf16<<<1024, 256, 0, stream>>>(W_proj, wproj_bf, C * C);

  // kqv = x @ W_kqv + b_kqv   (bf16 output, split order k|q|v)
  dim3 g1(N1 / 64, M / 128);
  gemm_bias_bf16<true><<<g1, 128, 0, stream>>>(x_bf, wkqv_bf, b_kqv, kqv_bf,
                                               M, N1, C);

  // flash attention with causal mask, scale 1/sqrt(T)
  dim3 g2(T / 16, B);
  attn_flash_bf16<<<g2, 512, 0, stream>>>(kqv_bf, ao_bf, T, C);

  // out = att_out @ W_proj + b_proj  (f32 output)
  dim3 g3(C / 64, M / 128);
  gemm_bias_bf16<false><<<g3, 128, 0, stream>>>(ao_bf, wproj_bf, b_proj, d_out,
                                                M, C, C);
}
